// ClipLoss_56564719289089
// MI455X (gfx1250) — matrix operand order repaired and checked by reading, not compiled, not run on hardware
//
#include <hip/hip_runtime.h>
#include <cstdint>
#include <cstddef>

typedef _Float16 v16h  __attribute__((ext_vector_type(16)));
typedef _Float16 f16x8 __attribute__((ext_vector_type(8)));
typedef float    v8f   __attribute__((ext_vector_type(8)));

#define N_ROWS 8192
#define DIM    512
#define EPS_C  0.001f

// ---------- reductions (wave32) ----------
__device__ __forceinline__ float reduce16(float x) {
    x += __shfl_xor(x, 1, 32);
    x += __shfl_xor(x, 2, 32);
    x += __shfl_xor(x, 4, 32);
    x += __shfl_xor(x, 8, 32);
    return x;
}
__device__ __forceinline__ float reduce32(float x) {
    x = reduce16(x);
    x += __shfl_xor(x, 16, 32);
    return x;
}

// ---------- fragment loaders ----------
// A (16-bit 16x32): lane&15 = row M; half-wave 0 holds K {0..7, 16..23},
// half-wave 1 holds K {8..15, 24..31}  -> two 16B chunks 32B apart.
__device__ __forceinline__ v16h load_afrag(const _Float16* p) {
    union { v16h v; f16x8 h[2]; } u;
    u.h[0] = *(const f16x8*)(p);
    u.h[1] = *(const f16x8*)(p + 16);
    return u.v;
}
// B (16-bit 32x16): lane&15 = column N; half-wave 0 holds K {0..7, 16..23},
// half-wave 1 holds K {8..15, 24..31} -> two 16B chunks 32B apart.
__device__ __forceinline__ v16h load_bfrag(const _Float16* p) {
    union { v16h v; f16x8 h[2]; } u;
    u.h[0] = *(const f16x8*)(p);
    u.h[1] = *(const f16x8*)(p + 16);
    return u.v;
}

// ---------- kernel 0: zero accumulators ----------
__global__ void clip_init_kernel(float* __restrict__ S, float* __restrict__ out) {
    int i = blockIdx.x * blockDim.x + threadIdx.x;
    if (i < N_ROWS) S[i] = 0.0f;
    if (i == 0)     out[0] = 0.0f;
}

// ---------- kernel 1: L2-normalize rows, emit f16 + inverse norms ----------
__global__ void clip_normalize_kernel(const float* __restrict__ v1,
                                      const float* __restrict__ v2,
                                      _Float16* __restrict__ v1h,
                                      _Float16* __restrict__ v2h,
                                      float* __restrict__ inv1,
                                      float* __restrict__ inv2) {
    int gwave = (int)((blockIdx.x * blockDim.x + threadIdx.x) >> 5);
    int lane  = (int)(threadIdx.x & 31);

    const float* src; _Float16* dst; float* invout; int row;
    if (gwave < N_ROWS) { src = v1; dst = v1h; invout = inv1; row = gwave; }
    else                { src = v2; dst = v2h; invout = inv2; row = gwave - N_ROWS; }

    const float4* p = (const float4*)(src + (size_t)row * DIM + lane * 16);
    float4 a = p[0], b = p[1], c = p[2], d = p[3];

    float ss = a.x*a.x + a.y*a.y + a.z*a.z + a.w*a.w
             + b.x*b.x + b.y*b.y + b.z*b.z + b.w*b.w
             + c.x*c.x + c.y*c.y + c.z*c.z + c.w*c.w
             + d.x*d.x + d.y*d.y + d.z*d.z + d.w*d.w;
    ss = reduce32(ss);
    float inv = rsqrtf(ss);
    if (lane == 0) invout[row] = inv;

    f16x8 h0, h1;
    h0[0] = (_Float16)(a.x*inv); h0[1] = (_Float16)(a.y*inv);
    h0[2] = (_Float16)(a.z*inv); h0[3] = (_Float16)(a.w*inv);
    h0[4] = (_Float16)(b.x*inv); h0[5] = (_Float16)(b.y*inv);
    h0[6] = (_Float16)(b.z*inv); h0[7] = (_Float16)(b.w*inv);
    h1[0] = (_Float16)(c.x*inv); h1[1] = (_Float16)(c.y*inv);
    h1[2] = (_Float16)(c.z*inv); h1[3] = (_Float16)(c.w*inv);
    h1[4] = (_Float16)(d.x*inv); h1[5] = (_Float16)(d.y*inv);
    h1[6] = (_Float16)(d.z*inv); h1[7] = (_Float16)(d.w*inv);

    f16x8* q = (f16x8*)(dst + (size_t)row * DIM + lane * 16);
    q[0] = h0;
    q[1] = h1;
}

// ---------- kernel 2: fused exp-sum GEMM on WMMA ----------
// Workgroup = 8 waves, owns 32 rows x 512 cols. Each wave owns 32 rows x 64
// cols: 2 row-tiles x 4 col-tiles = 8 accumulators, software-pipelined with
// double-buffered A/B fragments so WMMAs never wait on same-step loads.
// grid = (N/512 col-slabs, N/32 row-blocks).
__global__ void __launch_bounds__(256)
clip_gemm_expsum_kernel(const _Float16* __restrict__ v1h,
                        const _Float16* __restrict__ v2h,
                        const float* __restrict__ t,
                        float* __restrict__ S) {
    __shared__ float rowsum[32];

    const int lane    = (int)(threadIdx.x & 31);
    const int wave    = (int)(threadIdx.x >> 5);          // 0..7
    const int rowBase = (int)blockIdx.y * 32;
    const int colBase = (int)blockIdx.x * 512 + wave * 64;

    if (threadIdx.x < 32) rowsum[threadIdx.x] = 0.0f;
    __syncthreads();

    const float scale = __expf(t[0]);

    const _Float16* arow0 = v1h + (size_t)(rowBase + (lane & 15)) * DIM
                                + (lane >> 4) * 8;
    const _Float16* arow1 = arow0 + 16 * DIM;
    const _Float16* bcol  = v2h + (size_t)(colBase + (lane & 15)) * DIM
                                + (lane >> 4) * 8;

    v8f  acc[2][4] = {};
    v16h a[2][2];
    v16h b[2][4];

    // prologue: fragments for k = 0
    a[0][0] = load_afrag(arow0);
    a[0][1] = load_afrag(arow1);
    #pragma unroll
    for (int ct = 0; ct < 4; ++ct)
        b[0][ct] = load_bfrag(bcol + (size_t)ct * 16 * DIM);

    constexpr int NK = DIM / 32;   // 16 k-steps
    #pragma unroll
    for (int kk = 0; kk < NK; ++kk) {
        const int cur = kk & 1;
        const int nxt = cur ^ 1;
        const int kn  = (kk + 1) * 32;

        if (kk + 1 < NK) {
            a[nxt][0] = load_afrag(arow0 + kn);
            a[nxt][1] = load_afrag(arow1 + kn);
            #pragma unroll
            for (int ct = 0; ct < 4; ++ct)
                b[nxt][ct] = load_bfrag(bcol + (size_t)ct * 16 * DIM + kn);
        }

        #pragma unroll
        for (int rt = 0; rt < 2; ++rt) {
            #pragma unroll
            for (int ct = 0; ct < 4; ++ct) {
                acc[rt][ct] = __builtin_amdgcn_wmma_f32_16x16x32_f16(
                    false, a[cur][rt], false, b[cur][ct],
                    (short)0, acc[rt][ct], false, false);
            }
        }
    }

    // C/D layout: VGPR r -> row (r + 8*(lane>=16)), column (lane & 15).
    #pragma unroll
    for (int rt = 0; rt < 2; ++rt) {
        #pragma unroll
        for (int r = 0; r < 8; ++r) {
            float e = 0.0f;
            #pragma unroll
            for (int ct = 0; ct < 4; ++ct)
                e += __expf(acc[rt][ct][r] * scale);
            e = reduce16(e);
            if ((lane & 15) == 0) {
                int rloc = rt * 16 + r + ((lane >> 4) << 3);
                atomicAdd(&rowsum[rloc], e);   // ds_add_f32 across 8 waves
            }
        }
    }
    __syncthreads();
    if (threadIdx.x < 32)
        atomicAdd(&S[rowBase + threadIdx.x], rowsum[threadIdx.x]);
}

// ---------- kernel 3: fp32 diagonal + log + global reduction ----------
__global__ void clip_finalize_kernel(const float* __restrict__ v1,
                                     const float* __restrict__ v2,
                                     const float* __restrict__ inv1,
                                     const float* __restrict__ inv2,
                                     const float* __restrict__ t,
                                     const float* __restrict__ S,
                                     float* __restrict__ out) {
    int row  = (int)((blockIdx.x * blockDim.x + threadIdx.x) >> 5);
    int lane = (int)(threadIdx.x & 31);
    if (row >= N_ROWS) return;

    const float4* p1 = (const float4*)(v1 + (size_t)row * DIM + lane * 16);
    const float4* p2 = (const float4*)(v2 + (size_t)row * DIM + lane * 16);

    float dot = 0.0f;
    #pragma unroll
    for (int j = 0; j < 4; ++j) {
        float4 a = p1[j], b = p2[j];
        dot += a.x*b.x + a.y*b.y + a.z*b.z + a.w*b.w;
    }
    dot = reduce32(dot);

    if (lane == 0) {
        float scale = __expf(t[0]);
        float lii   = dot * inv1[row] * inv2[row] * scale;
        float per   = logf(S[row] + EPS_C) - lii;
        atomicAdd(out, per * (0.5f / (float)N_ROWS));
    }
}

extern "C" void kernel_launch(void* const* d_in, const int* in_sizes, int n_in,
                              void* d_out, int out_size, void* d_ws, size_t ws_size,
                              hipStream_t stream) {
    const float* v1 = (const float*)d_in[0];
    const float* v2 = (const float*)d_in[1];
    const float* t  = (const float*)d_in[2];
    float* out = (float*)d_out;

    char* ws = (char*)d_ws;
    const size_t half_bytes = (size_t)N_ROWS * DIM * sizeof(_Float16); // 8 MB each
    _Float16* v1h  = (_Float16*)ws;
    _Float16* v2h  = (_Float16*)(ws + half_bytes);
    float*    S    = (float*)(ws + 2 * half_bytes);
    float*    inv1 = S + N_ROWS;
    float*    inv2 = inv1 + N_ROWS;

    // 0) zero accumulators (output + row sums) — required every call.
    clip_init_kernel<<<(N_ROWS + 255) / 256, 256, 0, stream>>>(S, out);

    // 1) normalize both matrices: 2N waves, 8 waves per block.
    clip_normalize_kernel<<<(2 * N_ROWS) / 8, 256, 0, stream>>>(
        v1, v2, v1h, v2h, inv1, inv2);

    // 2) fused exp-sum GEMM: grid (col slabs of 512, row blocks of 32).
    dim3 ggrid(N_ROWS / 512, N_ROWS / 32);
    clip_gemm_expsum_kernel<<<ggrid, 256, 0, stream>>>(v1h, v2h, t, S);

    // 3) finalize: N waves, 8 per block.
    clip_finalize_kernel<<<N_ROWS / 8, 256, 0, stream>>>(
        v1, v2, inv1, inv2, t, S, out);
}
